// LinearSystem_77790447665177
// MI455X (gfx1250) — compile-verified, matrix operand order checked by reading, not run on hardware
//
#include <hip/hip_runtime.h>
#include <hip/hip_bf16.h>

typedef __attribute__((ext_vector_type(16))) __bf16 v16bf;
typedef __attribute__((ext_vector_type(8)))  __bf16 v8bf;
typedef __attribute__((ext_vector_type(8)))  float  v8f;

constexpr int BATCH = 64;
constexpr int NSTEP = 1024;
constexpr int XD    = 512;
constexpr int UD    = 256;
constexpr int YD    = 256;
constexpr long MROWS = (long)BATCH * NSTEP;   // 65536 flattened (b,step) rows

// ---------------- helpers ----------------

__device__ __forceinline__ __bf16 f2bf(float f) {
    unsigned u = __builtin_bit_cast(unsigned, f);
    u += 0x7FFFu + ((u >> 16) & 1u);               // round-to-nearest-even
    unsigned short s = (unsigned short)(u >> 16);
    return __builtin_bit_cast(__bf16, s);
}

#define WMMA_BF16(AF, BF, ACC) \
    __builtin_amdgcn_wmma_f32_16x16x32_bf16(false, (AF), false, (BF), (short)0, (ACC), false, false)

// Swizzled operand storage: a row-major [rows][K] fp32 matrix is repacked so
// that for each (tile16, ktile32) block, lane l's 16 operand half-words are
// contiguous (32B). Per ISA 7.12.2 (16-bit A/B operands):
//   m/n = tile*16 + (l&15); K = ktile*32 + (l>>4)*8 + {e&7} + 16*(e>>3)
__device__ __forceinline__ v16bf load_frag(const __bf16* __restrict__ base,
                                           long tile, int kt, int ktiles, int lane) {
    return *(const v16bf*)(base + (((tile * ktiles + kt) << 5) + lane) * 16);
}

// ---------------- kernels ----------------

// Convert fp32 [rows][K] -> swizzled bf16 operand buffer.
__global__ void swizzle_bf16_kernel(const float* __restrict__ src,
                                    __bf16* __restrict__ dst,
                                    int K, long nruns) {
    long run = (long)blockIdx.x * blockDim.x + threadIdx.x;
    if (run >= nruns) return;
    const int  ktiles = K >> 5;
    const int  h    = (int)(run & 1);              // 0: K+0..7  1: K+16..23
    const int  lane = (int)((run >> 1) & 31);
    const long blk  = run >> 6;                    // tile*ktiles + kt
    const int  kt   = (int)(blk % ktiles);
    const long tile = blk / ktiles;
    const long m    = (tile << 4) + (lane & 15);
    const int  k0   = (kt << 5) + ((lane >> 4) << 3) + (h << 4);
    const float* p  = src + m * (long)K + k0;
    float4 f0 = *(const float4*)(p);
    float4 f1 = *(const float4*)(p + 4);
    v8bf o;
    o[0] = f2bf(f0.x); o[1] = f2bf(f0.y); o[2] = f2bf(f0.z); o[3] = f2bf(f0.w);
    o[4] = f2bf(f1.x); o[5] = f2bf(f1.y); o[6] = f2bf(f1.z); o[7] = f2bf(f1.w);
    *(v8bf*)(dst + (blk << 9) + (lane << 4) + (h << 3)) = o;
}

// drive = u . b^T + noise_x . lx^T  -> x output region.
// Each wave computes an N-tile PAIR {np, np+16} for one M-tile: every
// A-fragment is loaded once and feeds two WMMAs.
__global__ __launch_bounds__(256) void drive_kernel(
    const __bf16* __restrict__ u_sw, const __bf16* __restrict__ nx_sw,
    const __bf16* __restrict__ b_sw, const __bf16* __restrict__ lx_sw,
    float* __restrict__ drv) {
    const int  lane = threadIdx.x & 31;
    const int  wv   = threadIdx.x >> 5;
    const long tile = (long)blockIdx.x * 8 + wv;   // (MROWS/16)*(32/2) tiles
    const int  np   = (int)(tile & 15);            // 16 N-tile pairs
    const long mt   = tile >> 4;
    v8f accA = {}, accB = {};
#pragma unroll 4
    for (int kt = 0; kt < UD / 32; ++kt) {
        v16bf af = load_frag(u_sw, mt, kt, UD / 32, lane);
        accA = WMMA_BF16(af, load_frag(b_sw, np,      kt, UD / 32, lane), accA);
        accB = WMMA_BF16(af, load_frag(b_sw, np + 16, kt, UD / 32, lane), accB);
    }
#pragma unroll 4
    for (int kt = 0; kt < XD / 32; ++kt) {
        v16bf af = load_frag(nx_sw, mt, kt, XD / 32, lane);
        accA = WMMA_BF16(af, load_frag(lx_sw, np,      kt, XD / 32, lane), accA);
        accB = WMMA_BF16(af, load_frag(lx_sw, np + 16, kt, XD / 32, lane), accB);
    }
    const long mrow = (mt << 4) + ((lane >> 4) << 3);
    const int  colA = (np << 4) + (lane & 15);
#pragma unroll
    for (int r = 0; r < 8; ++r) {
        drv[(mrow + r) * (long)XD + colA]       = accA[r];
        drv[(mrow + r) * (long)XD + colA + 256] = accB[r];
    }
}

// Sequential recurrence x_k = x_{k-1} a^T + drive_k, in place over the x
// region (pre-filled with drive). 4 blocks x 16 batches; 16x512 state double-
// buffered in LDS in swizzled operand layout. 16 waves: wave w owns column
// tiles {w, w+16}; B-fragments of tile w are REGISTER-CACHED for all 1024
// steps (128 VGPRs), tile w+16 streams from L2. 32 WMMAs + 1 barrier / step.
__global__ __launch_bounds__(512) void scan_kernel(
    const float* __restrict__ x0, const __bf16* __restrict__ a_sw,
    float* __restrict__ x, __bf16* __restrict__ x_sw) {
    alignas(32) __shared__ __bf16 xs[2][16][32][16];   // [buf][kt][lane][e] = 32KB
    const int tid  = threadIdx.x;
    const int lane = tid & 31;
    const int wv   = tid >> 5;                         // 0..15
    const int g    = blockIdx.x;                       // batches 16g .. 16g+15

    // load + swizzle x_0 into xs[0]: 1024 8-element runs, 2 per thread
    for (int run = tid; run < 1024; run += 512) {
        const int h  = run & 1;
        const int ln = (run >> 1) & 31;
        const int kt = run >> 6;                       // 0..15
        const int k0 = (kt << 5) + ((ln >> 4) << 3) + (h << 4);
        const float* p = x0 + (size_t)(16 * g + (ln & 15)) * XD + k0;
        float4 f0 = *(const float4*)(p);
        float4 f1 = *(const float4*)(p + 4);
        v8bf o;
        o[0] = f2bf(f0.x); o[1] = f2bf(f0.y); o[2] = f2bf(f0.z); o[3] = f2bf(f0.w);
        o[4] = f2bf(f1.x); o[5] = f2bf(f1.y); o[6] = f2bf(f1.z); o[7] = f2bf(f1.w);
        *(v8bf*)(&xs[0][kt][ln][h << 3]) = o;
    }
    __syncthreads();

    // register-cache the a^T fragments of column-tile wv (whole scan)
    v16bf bw[16];
#pragma unroll
    for (int kt = 0; kt < 16; ++kt) bw[kt] = load_frag(a_sw, wv, kt, 16, lane);

    const int mrow = (lane >> 4) << 3;
    const int colA = (wv << 4) + (lane & 15);          // 0..255
    // per-lane-constant LDS/x_sw scatter coords; colB = colA+256 shares all
    // low bits, so only kt differs (+8)
    const int w_ktA = colA >> 5;                       // 0..7
    const int w_lhi = (colA >> 3) & 1;
    const int w_e   = (colA & 7) | (((colA >> 4) & 1) << 3);
    int cur = 0;
    for (int step = 0; step < NSTEP; ++step) {
        v8f accA = {}, accB = {};
#pragma unroll
        for (int kt = 0; kt < 16; ++kt) {
            v16bf af = *(const v16bf*)(&xs[cur][kt][lane][0]);
            accA = WMMA_BF16(af, bw[kt], accA);
            accB = WMMA_BF16(af, load_frag(a_sw, wv + 16, kt, 16, lane), accB);
        }
#pragma unroll
        for (int r = 0; r < 8; ++r) {
            const int  m  = mrow + r;                  // batch within group
            const long b  = 16 * g + m;
            const size_t row = (size_t)b * NSTEP + step;
            // column colA (cached tile) and colA+256 (streamed tile)
            const float vA = accA[r] + x[row * XD + colA];
            const float vB = accB[r] + x[row * XD + colA + 256];
            x[row * XD + colA]       = vA;
            x[row * XD + colA + 256] = vB;
            const __bf16 vbA = f2bf(vA), vbB = f2bf(vB);
            xs[cur ^ 1][w_ktA]     [m | (w_lhi << 4)][w_e] = vbA;
            xs[cur ^ 1][w_ktA + 8] [m | (w_lhi << 4)][w_e] = vbB;
            // swizzled x for the y-GEMM: flat row = b*NSTEP + step
            const long xmt = (b << 6) + (step >> 4);
            const int  xl  = (step & 15) | (w_lhi << 4);
            x_sw[(((xmt << 4) + w_ktA)     << 9) + (xl << 4) + w_e] = vbA;
            x_sw[(((xmt << 4) + w_ktA + 8) << 9) + (xl << 4) + w_e] = vbB;
        }
        __syncthreads();                               // publish x_k
        cur ^= 1;
    }
}

// y = x . c^T + noise_y . ly^T ; N-tile pair {np, np+8} per wave.
__global__ __launch_bounds__(256) void y_kernel(
    const __bf16* __restrict__ x_sw, const __bf16* __restrict__ ny_sw,
    const __bf16* __restrict__ c_sw, const __bf16* __restrict__ ly_sw,
    float* __restrict__ y) {
    const int  lane = threadIdx.x & 31;
    const int  wv   = threadIdx.x >> 5;
    const long tile = (long)blockIdx.x * 8 + wv;   // (MROWS/16)*(16/2) tiles
    const int  np   = (int)(tile & 7);             // 8 N-tile pairs
    const long mt   = tile >> 3;
    v8f accA = {}, accB = {};
#pragma unroll 4
    for (int kt = 0; kt < XD / 32; ++kt) {
        v16bf af = load_frag(x_sw, mt, kt, XD / 32, lane);
        accA = WMMA_BF16(af, load_frag(c_sw, np,     kt, XD / 32, lane), accA);
        accB = WMMA_BF16(af, load_frag(c_sw, np + 8, kt, XD / 32, lane), accB);
    }
#pragma unroll 4
    for (int kt = 0; kt < YD / 32; ++kt) {
        v16bf af = load_frag(ny_sw, mt, kt, YD / 32, lane);
        accA = WMMA_BF16(af, load_frag(ly_sw, np,     kt, YD / 32, lane), accA);
        accB = WMMA_BF16(af, load_frag(ly_sw, np + 8, kt, YD / 32, lane), accB);
    }
    const long mrow = (mt << 4) + ((lane >> 4) << 3);
    const int  colA = (np << 4) + (lane & 15);
#pragma unroll
    for (int r = 0; r < 8; ++r) {
        y[(mrow + r) * (long)YD + colA]       = accA[r];
        y[(mrow + r) * (long)YD + colA + 128] = accB[r];
    }
}

// ---------------- host ----------------

extern "C" void kernel_launch(void* const* d_in, const int* in_sizes, int n_in,
                              void* d_out, int out_size, void* d_ws, size_t ws_size,
                              hipStream_t stream) {
    (void)in_sizes; (void)n_in; (void)out_size; (void)ws_size;
    const float* x0 = (const float*)d_in[0];
    const float* u  = (const float*)d_in[1];
    const float* nx = (const float*)d_in[2];
    const float* ny = (const float*)d_in[3];
    const float* a  = (const float*)d_in[4];
    const float* b  = (const float*)d_in[5];
    const float* c  = (const float*)d_in[6];
    const float* lx = (const float*)d_in[7];
    const float* ly = (const float*)d_in[8];

    float* xout = (float*)d_out;                        // [64][1024][512]
    float* yout = xout + MROWS * (long)XD;              // [64][1024][256]

    // workspace layout (bf16 elements), all offsets 32B-aligned
    __bf16* b_sw  = (__bf16*)d_ws;                      // 512x256
    __bf16* lx_sw = b_sw  + (long)XD * UD;              // 512x512
    __bf16* a_sw  = lx_sw + (long)XD * XD;              // 512x512
    __bf16* c_sw  = a_sw  + (long)XD * XD;              // 256x512
    __bf16* ly_sw = c_sw  + (long)YD * XD;              // 256x256
    __bf16* u_sw  = ly_sw + (long)YD * YD;              // 65536x256
    __bf16* nx_sw = u_sw  + MROWS * UD;                 // 65536x512
    __bf16* ny_sw = nx_sw + MROWS * XD;                 // 65536x256
    __bf16* x_sw  = ny_sw + MROWS * YD;                 // 65536x512  (~203 MB total)

    auto swz = [&](const float* s, __bf16* d, long rows, int K) {
        long nruns = rows * K / 8;
        swizzle_bf16_kernel<<<(int)((nruns + 255) / 256), 256, 0, stream>>>(s, d, K, nruns);
    };
    swz(b,  b_sw,  XD, UD);
    swz(lx, lx_sw, XD, XD);
    swz(a,  a_sw,  XD, XD);
    swz(c,  c_sw,  YD, XD);
    swz(ly, ly_sw, YD, YD);
    swz(u,  u_sw,  MROWS, UD);
    swz(nx, nx_sw, MROWS, XD);
    swz(ny, ny_sw, MROWS, YD);

    // drive -> x region: 4096 M-tiles x 16 N-tile-pairs, 8 waves/block
    drive_kernel<<<8192, 256, 0, stream>>>(u_sw, nx_sw, b_sw, lx_sw, xout);

    // sequential scan, in place: 4 batch-groups of 16, 16 waves each
    scan_kernel<<<4, 512, 0, stream>>>(x0, a_sw, xout, x_sw);

    // y: 4096 M-tiles x 8 N-tile-pairs, 8 waves/block
    y_kernel<<<4096, 256, 0, stream>>>(x_sw, ny_sw, c_sw, ly_sw, yout);
}
